// MySelfAttention_10170482557150
// MI455X (gfx1250) — matrix operand order repaired and checked by reading, not compiled, not run on hardware
//
#include <hip/hip_runtime.h>
#include <math.h>

// ---------------------------------------------------------------------------
// MI455X (gfx1250) self-attention, bf16 WMMA:
//   pass 1 (x3, templated): Q/K projection row-major bf16, V transposed bf16
//   pass 2: flash attention (online softmax), 16x128 out tile per wave,
//           K/V blocks staged to LDS via GLOBAL_LOAD_ASYNC_TO_LDS_B128
// ---------------------------------------------------------------------------

typedef __bf16 bf16;
typedef __attribute__((ext_vector_type(16))) __bf16 v16bf;
typedef __attribute__((ext_vector_type(8)))  float  v8f;

static constexpr int   DIM   = 1024;
static constexpr int   NTOK  = 4096;
static constexpr int   NB    = 4;
static constexpr int   ROWS  = NB * NTOK;      // 16384
static constexpr float NORM_FACT = 0.03125f;   // 1/sqrt(1024)

union Frag {
    v16bf v;
    uint4 q[2];   // two 16-byte halves
};

#define WMMA_BF16(A, Bm, C) \
    __builtin_amdgcn_wmma_f32_16x16x32_bf16(false, (A), false, (Bm), (short)0, (C), false, false)

// Async global->LDS DMA (CDNA5): per-lane 16B copy, tracked by ASYNCcnt.
static __device__ __forceinline__ void async_copy_b128(void* lds, const void* glob)
{
    const unsigned int loff = (unsigned int)(uintptr_t)lds;  // LDS byte offset
    asm volatile("global_load_async_to_lds_b128 %0, %1, off"
                 :: "v"(loff), "v"(glob) : "memory");
}
static __device__ __forceinline__ void wait_async_zero()
{
    asm volatile("s_wait_asynccnt 0x0" ::: "memory");
}

// ---------------------------------------------------------------------------
// Kernel 1 (Z = 0:Q, 1:K, 2:V-transposed):  y = x @ W^T, 128x128 tile/block.
// (fp32->bf16 conversion must pass through VGPRs, so staging stays synchronous)
// ---------------------------------------------------------------------------
template <int Z>
__global__ __launch_bounds__(256) void qkv_kernel(
    const float* __restrict__ x,
    const float* __restrict__ W,
    bf16* __restrict__ dst)          // Q/K: [ROWS][DIM]; V: Vt [DIM][ROWS]
{
    __shared__ __align__(16) bf16 sX[128 * 32];
    __shared__ __align__(16) bf16 sW[128 * 32];
    __shared__ __align__(16) bf16 sOut[128 * 128];

    const int mbase = blockIdx.y * 128;
    const int nbase = blockIdx.x * 128;
    const int tid   = threadIdx.x;
    const int wv    = tid >> 5;
    const int lane  = tid & 31;
    const int lm    = lane & 15;
    const int lh    = lane >> 4;

    v8f c[8];
    for (int i = 0; i < 8; i++)
        for (int j = 0; j < 8; j++) c[i][j] = 0.0f;

    for (int kk = 0; kk < DIM; kk += 32) {
#pragma unroll
        for (int j = 0; j < 4; j++) {
            const int c4 = tid + 256 * j;
            const int m  = c4 >> 3;
            const int k  = (c4 & 7) * 4;
            {
                const float4 f = *(const float4*)&x[(size_t)(mbase + m) * DIM + kk + k];
                bf16 h[4] = { (bf16)f.x, (bf16)f.y, (bf16)f.z, (bf16)f.w };
                *(uint2*)&sX[m * 32 + k] = *(const uint2*)h;
            }
            {
                const float4 f = *(const float4*)&W[(size_t)(nbase + m) * DIM + kk + k];
                bf16 h[4] = { (bf16)f.x, (bf16)f.y, (bf16)f.z, (bf16)f.w };
                *(uint2*)&sW[m * 32 + k] = *(const uint2*)h;
            }
        }
        __syncthreads();

        Frag a;
        {
            const bf16* ap = &sX[(wv * 16 + lm) * 32 + lh * 8];
            a.q[0] = *(const uint4*)ap;
            a.q[1] = *(const uint4*)(ap + 16);
        }
#pragma unroll
        for (int g4 = 0; g4 < 2; g4++) {
            Frag bfr[4];
#pragma unroll
            for (int i = 0; i < 4; i++) {
                const bf16* bp = &sW[((g4 * 4 + i) * 16 + lm) * 32 + lh * 8];
                bfr[i].q[0] = *(const uint4*)bp;
                bfr[i].q[1] = *(const uint4*)(bp + 16);
            }
#pragma unroll
            for (int i = 0; i < 4; i++)
                c[g4 * 4 + i] = WMMA_BF16(a.v, bfr[i].v, c[g4 * 4 + i]);
        }
        __syncthreads();
    }

    // ---- epilogue: C frags -> LDS (bf16) -> coalesced b128 global stores ----
#pragma unroll
    for (int nt = 0; nt < 8; nt++) {
#pragma unroll
        for (int r = 0; r < 8; r++) {
            const int ml = wv * 16 + r + 8 * lh;
            const int nl = nt * 16 + lm;
            if (Z == 2) sOut[nl * 128 + ml] = (bf16)c[nt][r];   // transposed
            else        sOut[ml * 128 + nl] = (bf16)c[nt][r];
        }
    }
    __syncthreads();

#pragma unroll
    for (int j = 0; j < 8; j++) {
        const int ch = tid + 256 * j;
        const int a0 = ch >> 4;
        const int e0 = (ch & 15) * 8;
        const uint4 val = *(const uint4*)&sOut[a0 * 128 + e0];
        if (Z == 2)
            *(uint4*)&dst[(size_t)(nbase + a0) * ROWS + mbase + e0] = val;
        else
            *(uint4*)&dst[(size_t)(mbase + a0) * DIM + nbase + e0] = val;
    }
}

// ---------------------------------------------------------------------------
// Kernel 2: flash attention.
// grid = (8 v-chunks of 128 dims, 32 q-blocks of 128 rows, 4 batches).
// wave w owns q rows [qblk*128+16w, +16) x dims [vchunk*128, +128).
// K (per 128-dim chunk) and Vt blocks are staged into an aliased 32 KB LDS
// buffer by the async-to-LDS DMA path (ASYNCcnt), shared by all 8 waves.
// ---------------------------------------------------------------------------
__global__ __launch_bounds__(256) void attn_kernel(
    const bf16* __restrict__ Q,
    const bf16* __restrict__ K,
    const bf16* __restrict__ Vt,
    float* __restrict__ out)
{
    __shared__ __align__(16) bf16 sStage[128 * 128];  // K chunk / Vt block (aliased)
    __shared__ __align__(16) bf16 sP[8][16 * 128];    // per-wave P tile

    const int vchunk = blockIdx.x;   // 0..7
    const int qblk   = blockIdx.y;   // 0..31
    const int b      = blockIdx.z;   // 0..3
    const int tid    = threadIdx.x;
    const int wv     = tid >> 5;
    const int lane   = tid & 31;
    const int lm     = lane & 15;
    const int lh     = lane >> 4;

    const int qrow = b * NTOK + qblk * 128 + wv * 16;
    const bf16* aq = Q + (size_t)(qrow + lm) * DIM;

    v8f o[8];
    for (int i = 0; i < 8; i++)
        for (int j = 0; j < 8; j++) o[i][j] = 0.0f;
    float mrow[8], lrow[8];
    for (int r = 0; r < 8; r++) { mrow[r] = -INFINITY; lrow[r] = 0.0f; }

    for (int kb = 0; kb < 32; kb++) {
        const int keyb = b * NTOK + kb * 128;

        // hint: pull next key block toward the caches (global_prefetch_b8)
        if (kb < 31)
            __builtin_prefetch(K + (size_t)(keyb + 128 + lane * 4) * DIM, 0, 1);

        // ---- S = Q @ K^T, streaming 128-dim chunks of K through LDS ----
        v8f s[8];
        for (int i = 0; i < 8; i++)
            for (int j = 0; j < 8; j++) s[i][j] = 0.0f;

        for (int kc = 0; kc < DIM; kc += 128) {
            __syncthreads();   // previous sStage readers done
            // async stage K[keyb..+128)[kc..+128) -> sStage[key][dim]
#pragma unroll
            for (int j = 0; j < 8; j++) {
                const int ch  = tid + 256 * j;       // b128 chunk id (2048 total)
                const int key = ch >> 4;
                const int e0  = (ch & 15) * 8;
                async_copy_b128(&sStage[key * 128 + e0],
                                &K[(size_t)(keyb + key) * DIM + kc + e0]);
            }
            wait_async_zero();
            __syncthreads();

#pragma unroll
            for (int k2 = 0; k2 < 128; k2 += 32) {
                Frag a;
                a.q[0] = *(const uint4*)(aq + kc + k2 + lh * 8);
                a.q[1] = *(const uint4*)(aq + kc + k2 + lh * 8 + 16);
#pragma unroll
                for (int g4 = 0; g4 < 2; g4++) {
                    Frag bfr[4];
#pragma unroll
                    for (int i = 0; i < 4; i++) {
                        const bf16* bp =
                            &sStage[((g4 * 4 + i) * 16 + lm) * 128 + k2 + lh * 8];
                        bfr[i].q[0] = *(const uint4*)bp;
                        bfr[i].q[1] = *(const uint4*)(bp + 16);
                    }
#pragma unroll
                    for (int i = 0; i < 4; i++)
                        s[g4 * 4 + i] = WMMA_BF16(a.v, bfr[i].v, s[g4 * 4 + i]);
                }
            }
        }

        // ---- online softmax update (row r + 8*lh) ----
#pragma unroll
        for (int r = 0; r < 8; r++) {
            float mx = s[0][r];
#pragma unroll
            for (int nt = 1; nt < 8; nt++) mx = fmaxf(mx, s[nt][r]);
            mx = fmaxf(mx, __shfl_xor(mx, 1, 32));
            mx = fmaxf(mx, __shfl_xor(mx, 2, 32));
            mx = fmaxf(mx, __shfl_xor(mx, 4, 32));
            mx = fmaxf(mx, __shfl_xor(mx, 8, 32));
            const float mnew  = fmaxf(mrow[r], mx);
            const float scale = __expf(mrow[r] - mnew);
            mrow[r] = mnew;

            float partial = 0.0f;
#pragma unroll
            for (int nt = 0; nt < 8; nt++) {
                const float p = __expf(s[nt][r] - mnew);
                partial += p;
                sP[wv][(r + 8 * lh) * 128 + nt * 16 + lm] = (bf16)p;
            }
            partial += __shfl_xor(partial, 1, 32);
            partial += __shfl_xor(partial, 2, 32);
            partial += __shfl_xor(partial, 4, 32);
            partial += __shfl_xor(partial, 8, 32);
            lrow[r] = lrow[r] * scale + partial;
#pragma unroll
            for (int nt = 0; nt < 8; nt++) o[nt][r] *= scale;
        }

        // ---- restage (async): sStage <- Vt[vchunk dims][key block] ----
        __syncthreads();   // sK-chunk readers done; sP written
#pragma unroll
        for (int j = 0; j < 8; j++) {
            const int ch = tid + 256 * j;
            const int d  = ch >> 4;              // 0..127 local dim
            const int e0 = (ch & 15) * 8;        // key offset
            async_copy_b128(&sStage[d * 128 + e0],
                            &Vt[(size_t)(vchunk * 128 + d) * ROWS + keyb + e0]);
        }
        wait_async_zero();
        __syncthreads();

        // ---- O += P @ V  (P and V both from LDS) ----
#pragma unroll
        for (int kp = 0; kp < 128; kp += 32) {
            Frag a;
            const bf16* pp = &sP[wv][lm * 128 + kp + lh * 8];
            a.q[0] = *(const uint4*)pp;
            a.q[1] = *(const uint4*)(pp + 16);
#pragma unroll
            for (int g4 = 0; g4 < 2; g4++) {
                Frag bfr[4];
#pragma unroll
                for (int i = 0; i < 4; i++) {
                    const bf16* bp =
                        &sStage[((g4 * 4 + i) * 16 + lm) * 128 + kp + lh * 8];
                    bfr[i].q[0] = *(const uint4*)bp;
                    bfr[i].q[1] = *(const uint4*)(bp + 16);
                }
#pragma unroll
                for (int i = 0; i < 4; i++)
                    o[g4 * 4 + i] = WMMA_BF16(a.v, bfr[i].v, o[g4 * 4 + i]);
            }
        }
    }

    // ---- att = softmax(S) * NORM_FACT @ V = O * NORM/l ----
#pragma unroll
    for (int r = 0; r < 8; r++) {
        const float fac = NORM_FACT / lrow[r];
        const int g = qrow + r + 8 * lh;
#pragma unroll
        for (int nt = 0; nt < 8; nt++) {
            const int d = vchunk * 128 + nt * 16 + lm;
            out[(size_t)g * DIM + d] = o[nt][r] * fac;
        }
    }
}

// ---------------------------------------------------------------------------
extern "C" void kernel_launch(void* const* d_in, const int* in_sizes, int n_in,
                              void* d_out, int out_size, void* d_ws, size_t ws_size,
                              hipStream_t stream)
{
    const float* x  = (const float*)d_in[0];
    const float* Wq = (const float*)d_in[1];
    const float* Wk = (const float*)d_in[2];
    const float* Wv = (const float*)d_in[3];

    // workspace: Q[16384*1024] bf16, K same, Vt[1024*16384] bf16 = 96 MB
    bf16* Q  = (bf16*)d_ws;
    bf16* K  = Q + (size_t)ROWS * DIM;
    bf16* Vt = K + (size_t)ROWS * DIM;
    float* out = (float*)d_out;

    dim3 g1(8, 128);
    qkv_kernel<0><<<g1, 256, 0, stream>>>(x, Wq, Q);
    qkv_kernel<1><<<g1, 256, 0, stream>>>(x, Wk, K);
    qkv_kernel<2><<<g1, 256, 0, stream>>>(x, Wv, Vt);

    attn_kernel<<<dim3(8, 32, 4), 256, 0, stream>>>(Q, K, Vt, out);
}
